// CachedAttention_63007170232491
// MI455X (gfx1250) — matrix operand order repaired and checked by reading, not compiled, not run on hardware
//
#include <hip/hip_runtime.h>
#include <math.h>

// Problem constants (from reference): B=4, S=2048, KV=2048, T=KV+S=4096, D=2048
#define BQ  4
#define SQ  2048
#define KVQ 2048
#define TQ  4096
#define DQ  2048

typedef __bf16 v16bf __attribute__((ext_vector_type(16)));
typedef __bf16 v4bf  __attribute__((ext_vector_type(4)));
typedef float  v8f   __attribute__((ext_vector_type(8)));

// ---------------------------------------------------------------------------
// CDNA5 async global->LDS copy (GLOBAL_LOAD_ASYNC_TO_LDS_B128, ASYNCcnt),
// with a portable fallback if the builtin is absent on this toolchain.
// Builtin signature (from hipcc diagnostic): (v4i AS1*, v4i AS3*, imm, imm)
// ---------------------------------------------------------------------------
#if defined(__gfx1250__) && __has_builtin(__builtin_amdgcn_global_load_async_to_lds_b128)
#define USE_ASYNC_LDS 1
#else
#define USE_ASYNC_LDS 0
#endif

typedef int v4i_ __attribute__((__vector_size__(4 * sizeof(int))));

__device__ __forceinline__ void cp16_g2l(void* lds, const void* g) {
#if USE_ASYNC_LDS
  typedef __attribute__((address_space(1))) v4i_* gptr_t;
  typedef __attribute__((address_space(3))) v4i_* lptr_t;
  __builtin_amdgcn_global_load_async_to_lds_b128((gptr_t)g, (lptr_t)lds, 0, 0);
#else
  *(float4*)lds = *(const float4*)g;
#endif
}

__device__ __forceinline__ void wait_async_lds() {
#if USE_ASYNC_LDS
#if __has_builtin(__builtin_amdgcn_s_wait_asynccnt)
  __builtin_amdgcn_s_wait_asynccnt(0);
#else
  asm volatile("s_wait_asynccnt 0" ::: "memory");
#endif
#endif
}

// ---------------------------------------------------------------------------
// WMMA fragment loaders (wave32; A/B use this code's contiguous k order, same for both)
//
// A (16x32, bf16): lane L -> row m = L&15 ; lanes 0-15 hold K = 0..15,
//                  lanes 16-31 hold K = 16..31; 2 values per VGPR.
// B (32x16, bf16): lane L -> col n = L&15 ; lanes 0-15 hold K = 0..15,
//                  lanes 16-31 hold K = 16..31 (contiguous halves).
// C/D (16x16,f32): lane L -> col n = L&15 ; VGPR v -> row v (+8 for lanes>=16)
// ---------------------------------------------------------------------------

__device__ __forceinline__ v16bf load_a_bf16(const __bf16* A, int ld, long row0, int k0) {
  const int lane = threadIdx.x & 31;
  const int m  = lane & 15;
  const int kb = (lane & 16) ? 16 : 0;
  const __bf16* p = A + (row0 + m) * (long)ld + k0;
  v16bf a;
#pragma unroll
  for (int j = 0; j < 8; ++j) {
    const int k = kb + 2 * j; // {0,2,...,14}+kb
    a[2 * j]     = p[k];
    a[2 * j + 1] = p[k + 1];
  }
  return a;
}

// B for NT gemm: math B[k][n] = Bsrc[n0+n][k0+k]  (Bsrc row-major [N][K]).
// The 16 K-values per lane are contiguous -> single aligned 32B vector load.
__device__ __forceinline__ v16bf load_bt_bf16(const __bf16* Bsrc, int ld, int n0, int k0) {
  const int lane = threadIdx.x & 31;
  const int n  = lane & 15;
  const int kb = (lane & 16) ? 16 : 0;
  return *(const v16bf*)(Bsrc + (long)(n0 + n) * ld + k0 + kb);
}

// B for NN gemm: math B[k][n] = Bsrc[k0+k][n0+n]  (Bsrc row-major [K][N])
__device__ __forceinline__ v16bf load_bn_bf16(const __bf16* Bsrc, int ld, int n0, int k0) {
  const int lane = threadIdx.x & 31;
  const int n  = lane & 15;
  const int kb = (lane & 16) ? 16 : 0;
  const __bf16* p = Bsrc + (long)(k0 + kb) * ld + n0 + n;
  v16bf b;
#pragma unroll
  for (int h = 0; h < 16; ++h) b[h] = p[(long)h * ld];
  return b;
}

__device__ __forceinline__ v8f wmma_bf16(v16bf a, v16bf b, v8f c) {
  return __builtin_amdgcn_wmma_f32_16x16x32_bf16(false, a, false, b,
                                                 (short)0, c, false, false);
}

// ---------------------------------------------------------------------------
// Kernel: elementwise fp32 -> bf16 (each element converted exactly once)
// ---------------------------------------------------------------------------
__global__ void ca_cast_bf16(const float4* __restrict__ src,
                             unsigned short* __restrict__ dst_raw, long n4) {
  __bf16* dst = (__bf16*)dst_raw;
  const long i = (long)blockIdx.x * blockDim.x + threadIdx.x;
  if (i >= n4) return;
  const float4 t = src[i];
  v4bf o = {(__bf16)t.x, (__bf16)t.y, (__bf16)t.z, (__bf16)t.w};
  *(v4bf*)(dst + i * 4) = o;
}

// ---------------------------------------------------------------------------
// Kernel 0: seed caches with past_k / past_v: fp32 copy into d_out caches AND
// bf16 mirror into ws (operands for the attention GEMMs).
// ---------------------------------------------------------------------------
__global__ void ca_seed_cache(const float4* __restrict__ pk,
                              const float4* __restrict__ pv,
                              float4* __restrict__ ck, float4* __restrict__ cv,
                              unsigned short* __restrict__ kb_raw,
                              unsigned short* __restrict__ vb_raw) {
  const long n4 = (long)BQ * KVQ * DQ / 4;
  long i = (long)blockIdx.x * blockDim.x + threadIdx.x;
  const float4* src;
  float4* dst;
  __bf16* bdst;
  if (i < n4) { src = pk; dst = ck; bdst = (__bf16*)kb_raw; }
  else        { src = pv; dst = cv; bdst = (__bf16*)vb_raw; i -= n4; }
  if (i >= n4) return;
  const long per = (long)KVQ * DQ / 4;
  const long b = i / per, r = i % per;
  const float4 t = src[b * per + r];
  dst[b * ((long)TQ * DQ / 4) + r] = t;
  v4bf o = {(__bf16)t.x, (__bf16)t.y, (__bf16)t.z, (__bf16)t.w};
  *(v4bf*)(bdst + b * (long)TQ * DQ + r * 4) = o;
}

// ---------------------------------------------------------------------------
// Kernel 1: Y = X @ W^T  (bf16 operands, NT gemm)
// Block tile 64(M) x 128(N), K-step 32; bf16 tiles async-staged in LDS.
// 8 waves in 2x4 grid, each a 32x32 register tile (4 WMMA / K-step).
// Epilogue: optional fp32 store (cache outputs) + bf16 store (ws operands).
// ---------------------------------------------------------------------------
__global__ void __launch_bounds__(256)
ca_proj(const unsigned short* __restrict__ X_raw, const unsigned short* __restrict__ W_raw,
        float* __restrict__ outF, long fBatch, long fRowOff,
        unsigned short* __restrict__ outB_raw, long bBatch, long bRowOff) {
  __shared__ __bf16 lA[64 * 32];    // 4 KB
  __shared__ __bf16 lB[128 * 32];   // 8 KB
  const __bf16* X = (const __bf16*)X_raw;
  const __bf16* W = (const __bf16*)W_raw;
  __bf16* outB = (__bf16*)outB_raw;

  const int tid  = threadIdx.x;
  const int wave = tid >> 5;
  const int wm   = wave >> 2;            // 0..1
  const int wn   = wave & 3;             // 0..3
  const long row0 = (long)(blockIdx.x >> 4) * 64;  // never crosses a batch
  const int  n0   = (blockIdx.x & 15) * 128;

  v8f z = {0.f, 0.f, 0.f, 0.f, 0.f, 0.f, 0.f, 0.f};
  v8f acc[2][2] = {{z, z}, {z, z}};

  for (int kk = 0; kk < DQ; kk += 32) {
    __syncthreads();                     // previous tile fully consumed
    // stage A: 64x32 bf16 = 256 x 16B ; 1 chunk/thread
    {
      const int r = tid >> 2, q = tid & 3;   // 4 x 16B per row
      cp16_g2l(&lA[r * 32 + q * 8], &X[(row0 + r) * (long)DQ + kk + q * 8]);
    }
    // stage B: 128x32 bf16 = 512 x 16B ; 2 chunks/thread
#pragma unroll
    for (int i = 0; i < 2; ++i) {
      const int c = tid + 256 * i;
      const int r = c >> 2, q = c & 3;
      cp16_g2l(&lB[r * 32 + q * 8], &W[(long)(n0 + r) * DQ + kk + q * 8]);
    }
    __builtin_prefetch(&X[row0 * DQ + kk + 32], 0, 0);      // next A tile
    __builtin_prefetch(&W[(long)n0 * DQ + kk + 32], 0, 0);  // next B tile
    wait_async_lds();
    __syncthreads();

    v16bf af[2], bf[2];
#pragma unroll
    for (int mi = 0; mi < 2; ++mi) af[mi] = load_a_bf16(lA, 32, wm * 32 + mi * 16, 0);
#pragma unroll
    for (int ni = 0; ni < 2; ++ni) bf[ni] = load_bt_bf16(lB, 32, wn * 32 + ni * 16, 0);
#pragma unroll
    for (int mi = 0; mi < 2; ++mi)
#pragma unroll
      for (int ni = 0; ni < 2; ++ni)
        acc[mi][ni] = wmma_bf16(af[mi], bf[ni], acc[mi][ni]);
  }

  const int lane = tid & 31;
  const int col  = lane & 15;
  const int rb   = (lane & 16) ? 8 : 0;
  const int bidx = (int)(row0 / SQ);
  const int s0   = (int)(row0 % SQ);
#pragma unroll
  for (int mi = 0; mi < 2; ++mi) {
#pragma unroll
    for (int ni = 0; ni < 2; ++ni) {
#pragma unroll
      for (int v = 0; v < 8; ++v) {
        const int  lr  = wm * 32 + mi * 16 + rb + v;
        const long c   = n0 + wn * 32 + ni * 16 + col;
        const float val = acc[mi][ni][v];
        if (outF)
          outF[(long)bidx * fBatch + (fRowOff + s0 + lr) * (long)DQ + c] = val;
        outB[(long)bidx * bBatch + (bRowOff + s0 + lr) * (long)DQ + c] = (__bf16)val;
      }
    }
  }
}

// ---------------------------------------------------------------------------
// Kernel 2: scores = (Q @ Kb^T) / sqrt(D), fused row softmax (all bf16 in).
// One block = one 16-row query strip; 16 x 4096 f32 score strip in 256 KB
// dynamic LDS (within CDNA5's 320 KB/WGP). P written to ws as bf16.
// ---------------------------------------------------------------------------
__global__ void ca_scores_softmax(const unsigned short* __restrict__ Qb_raw,
                                  const unsigned short* __restrict__ Kb_raw,
                                  unsigned short* __restrict__ P_raw) {
  extern __shared__ float sc[];   // [16][TQ]
  const __bf16* Qb = (const __bf16*)Qb_raw;
  const __bf16* Kb = (const __bf16*)Kb_raw;
  __bf16* P = (__bf16*)P_raw;

  const int blk  = blockIdx.x;          // B * S/16 = 512
  const int b    = blk >> 7;
  const int mt   = blk & 127;
  const int row0 = mt * 16;             // within batch
  const __bf16* Qbase = Qb + (long)b * SQ * DQ;
  const __bf16* Kbase = Kb + (long)b * TQ * DQ;
  const int wave = threadIdx.x >> 5;
  const int lane = threadIdx.x & 31;
  const float scale = 0.022097086912079608f;   // 1/sqrt(2048)

  for (int nt = wave; nt < TQ / 16; nt += 8) {
    const int t0 = nt * 16;
    v8f acc = {0.f, 0.f, 0.f, 0.f, 0.f, 0.f, 0.f, 0.f};
    for (int kk = 0; kk < DQ; kk += 32) {
      __builtin_prefetch(&Kbase[(long)t0 * DQ + kk + 32], 0, 0);
      v16bf a  = load_a_bf16(Qbase, DQ, row0, kk);
      v16bf bm = load_bt_bf16(Kbase, DQ, t0, kk);
      acc = wmma_bf16(a, bm, acc);
    }
    const int col = lane & 15;
    const int rb  = (lane & 16) ? 8 : 0;
#pragma unroll
    for (int v = 0; v < 8; ++v)
      sc[(rb + v) * TQ + t0 + col] = acc[v] * scale;
  }
  __syncthreads();

  // Row softmax: wave w owns rows 2w and 2w+1; wave32 shuffle reductions.
#pragma unroll
  for (int rr = 0; rr < 2; ++rr) {
    const int r = wave * 2 + rr;
    float* row = sc + (long)r * TQ;
    float mx = -3.402823466e+38f;
    for (int i = lane; i < TQ; i += 32) mx = fmaxf(mx, row[i]);
#pragma unroll
    for (int off = 16; off; off >>= 1) mx = fmaxf(mx, __shfl_xor(mx, off, 32));
    float sum = 0.f;
    for (int i = lane; i < TQ; i += 32) {
      float e = __expf(row[i] - mx);
      row[i] = e;
      sum += e;
    }
#pragma unroll
    for (int off = 16; off; off >>= 1) sum += __shfl_xor(sum, off, 32);
    const float inv = 1.0f / sum;
    __bf16* prow = P + ((long)b * SQ + row0 + r) * TQ;
    for (int i = lane; i < TQ; i += 32) prow[i] = (__bf16)(row[i] * inv);
  }
}

// ---------------------------------------------------------------------------
// Kernel 3: out = round((P @ Vb) * 1e4, RNE) * 1e-4   (NN gemm over T, bf16 in)
// ---------------------------------------------------------------------------
__global__ void __launch_bounds__(256)
ca_pv(const unsigned short* __restrict__ P_raw,
      const unsigned short* __restrict__ Vb_raw, float* __restrict__ O) {
  __shared__ __bf16 lA[64 * 32];    // 4 KB
  __shared__ __bf16 lB[32 * 128];   // 8 KB
  const __bf16* P  = (const __bf16*)P_raw;
  const __bf16* Vb = (const __bf16*)Vb_raw;

  const int tid  = threadIdx.x;
  const int wave = tid >> 5;
  const int wm   = wave >> 2;            // 0..1
  const int wn   = wave & 3;             // 0..3
  const long row0 = (long)(blockIdx.x >> 4) * 64;  // global row in [0, B*S)
  const int  n0   = (blockIdx.x & 15) * 128;
  const int  b    = (int)(row0 / SQ);
  const long srow = row0 % SQ;
  const __bf16* Pbase = P  + (long)b * SQ * TQ;
  const __bf16* Vbase = Vb + (long)b * TQ * DQ;

  v8f z = {0.f, 0.f, 0.f, 0.f, 0.f, 0.f, 0.f, 0.f};
  v8f acc[2][2] = {{z, z}, {z, z}};

  for (int kk = 0; kk < TQ; kk += 32) {
    __syncthreads();
    // stage A: 64x32 bf16 = 256 x 16B ; 1 chunk/thread
    {
      const int r = tid >> 2, q = tid & 3;
      cp16_g2l(&lA[r * 32 + q * 8], &Pbase[(srow + r) * (long)TQ + kk + q * 8]);
    }
    // stage B: 32x128 bf16 = 512 x 16B ; 2 chunks/thread
#pragma unroll
    for (int i = 0; i < 2; ++i) {
      const int c = tid + 256 * i;
      const int r = c >> 4, q = c & 15;  // 16 x 16B per row of 128 bf16
      cp16_g2l(&lB[r * 128 + q * 8], &Vbase[(long)(kk + r) * DQ + n0 + q * 8]);
    }
    __builtin_prefetch(&Vbase[(long)(kk + 32) * DQ + n0], 0, 0);
    wait_async_lds();
    __syncthreads();

    v16bf af[2], bf[2];
#pragma unroll
    for (int mi = 0; mi < 2; ++mi) af[mi] = load_a_bf16(lA, 32, wm * 32 + mi * 16, 0);
#pragma unroll
    for (int ni = 0; ni < 2; ++ni) bf[ni] = load_bn_bf16(lB, 128, wn * 32 + ni * 16, 0);
#pragma unroll
    for (int mi = 0; mi < 2; ++mi)
#pragma unroll
      for (int ni = 0; ni < 2; ++ni)
        acc[mi][ni] = wmma_bf16(af[mi], bf[ni], acc[mi][ni]);
  }

  const int lane = tid & 31;
  const int col  = lane & 15;
  const int rb   = (lane & 16) ? 8 : 0;
#pragma unroll
  for (int mi = 0; mi < 2; ++mi) {
#pragma unroll
    for (int ni = 0; ni < 2; ++ni) {
#pragma unroll
      for (int v = 0; v < 8; ++v) {
        const int  lr = wm * 32 + mi * 16 + rb + v;
        const long c  = n0 + wn * 32 + ni * 16 + col;
        // torch.round(x, decimals=4): round-half-even at 1e-4 (v_rndne_f32)
        O[(row0 + lr) * (long)DQ + c] = rintf(acc[mi][ni][v] * 1.0e4f) * 1.0e-4f;
      }
    }
  }
}

// ---------------------------------------------------------------------------
// d_out = [ out (B*S*D) | cache_k (B*T*D) | cache_v (B*T*D) ]  (f32)
// d_ws (bf16): Xb | Wqb | Wkb | Wvb | Qb | Kb | Vb | Pb  ~= 294 MB
// ---------------------------------------------------------------------------
extern "C" void kernel_launch(void* const* d_in, const int* in_sizes, int n_in,
                              void* d_out, int out_size, void* d_ws, size_t ws_size,
                              hipStream_t stream) {
  const float* x      = (const float*)d_in[0];
  const float* past_k = (const float*)d_in[1];
  const float* past_v = (const float*)d_in[2];
  const float* Wq     = (const float*)d_in[3];
  const float* Wk     = (const float*)d_in[4];
  const float* Wv     = (const float*)d_in[5];

  float* out = (float*)d_out;
  float* ck  = out + (long)BQ * SQ * DQ;
  float* cv  = ck  + (long)BQ * TQ * DQ;

  const size_t nX  = (size_t)BQ * SQ * DQ;   // 16,777,216
  const size_t nW  = (size_t)DQ * DQ;        //  4,194,304
  const size_t nKV = (size_t)BQ * TQ * DQ;   // 33,554,432
  unsigned short* Xb  = (unsigned short*)d_ws;
  unsigned short* Wqb = Xb  + nX;
  unsigned short* Wkb = Wqb + nW;
  unsigned short* Wvb = Wkb + nW;
  unsigned short* Qb  = Wvb + nW;
  unsigned short* Kb  = Qb  + nX;
  unsigned short* Vb  = Kb  + nKV;
  unsigned short* Pb  = Vb  + nKV;           // B*S*T bf16

  // 0) one-time fp32 -> bf16 operand conversion (read-once, write-once)
  ca_cast_bf16<<<(int)(nX / 4 / 256), 256, 0, stream>>>((const float4*)x, Xb, nX / 4);
  ca_cast_bf16<<<(int)(nW / 4 / 256), 256, 0, stream>>>((const float4*)Wq, Wqb, nW / 4);
  ca_cast_bf16<<<(int)(nW / 4 / 256), 256, 0, stream>>>((const float4*)Wk, Wkb, nW / 4);
  ca_cast_bf16<<<(int)(nW / 4 / 256), 256, 0, stream>>>((const float4*)Wv, Wvb, nW / 4);
  // 0b) seed fp32 caches + bf16 mirrors with past K/V
  ca_seed_cache<<<32768, 256, 0, stream>>>((const float4*)past_k, (const float4*)past_v,
                                           (float4*)ck, (float4*)cv, Kb, Vb);
  // 1) projections (bf16 GEMM): Q -> Qb; K -> ck + Kb; V -> cv + Vb
  ca_proj<<<2048, 256, 0, stream>>>(Xb, Wqb, nullptr, 0, 0,
                                    Qb, (long)SQ * DQ, 0);
  ca_proj<<<2048, 256, 0, stream>>>(Xb, Wkb, ck, (long)TQ * DQ, (long)KVQ,
                                    Kb, (long)TQ * DQ, (long)KVQ);
  ca_proj<<<2048, 256, 0, stream>>>(Xb, Wvb, cv, (long)TQ * DQ, (long)KVQ,
                                    Vb, (long)TQ * DQ, (long)KVQ);
  // 2) scores + softmax (256 KB dynamic LDS score strip per block)
  ca_scores_softmax<<<512, 256, 16 * TQ * sizeof(float), stream>>>(Qb, Kb, Pb);
  // 3) P @ V with round-to-1e-4 epilogue
  ca_pv<<<2048, 256, 0, stream>>>(Pb, Vb, out);
}
